// D4RTEncoder_56006373540207
// MI455X (gfx1250) — matrix operand order repaired and checked by reading, not compiled, not run on hardware
//
#include <hip/hip_runtime.h>
#include <cstdint>
#include <cstddef>

// ---------------------------------------------------------------- types
typedef _Float16 half_t;
typedef __attribute__((ext_vector_type(16))) _Float16 v16h;
typedef __attribute__((ext_vector_type(8)))  _Float16 h8;
typedef __attribute__((ext_vector_type(8)))  float    v8f;

static __device__ __forceinline__ v16h combine16(h8 lo, h8 hi) {
    return __builtin_shufflevector(lo, hi, 0,1,2,3,4,5,6,7,8,9,10,11,12,13,14,15);
}

// CDNA5 async global->LDS copy (ASYNCcnt-tracked, no VGPR round trip).
// vdst = LDS byte address, vaddr = 64-bit global address.
static __device__ __forceinline__ void async_load_b128(const half_t* g, half_t* l) {
    unsigned lds = (unsigned)(uintptr_t)l;   // low 32 bits of flat shared ptr = LDS addr
    asm volatile("global_load_async_to_lds_b128 %0, %1, off"
                 :: "v"(lds), "v"(g) : "memory");
}
static __device__ __forceinline__ void async_wait4() {
    asm volatile("s_wait_asynccnt 0x4" ::: "memory");
}
static __device__ __forceinline__ void async_wait0() {
    asm volatile("s_wait_asynccnt 0x0" ::: "memory");
}

#define CC   768
#define NHD  12
#define HD   64
#define MLPD 3072
#define BB   32
#define NTOK 256          // patch tokens per image
#define NALL 257          // + ar token
#define MTOK (BB*NTOK)    // 8192
#define MALL (BB*NALL)    // 8224
#define NP   288          // padded key count for global attention (mult of 32)

// ---------------------------------------------------------------- f32 -> f16 convert
__global__ void cvt_kernel(const float* __restrict__ in, half_t* __restrict__ out, long n) {
    long i = (long)blockIdx.x * 256 + threadIdx.x;
    if (i < n) out[i] = (half_t)in[i];
}

// ---------------------------------------------------------------- im2col for 16x16 patch conv
__global__ void im2col_kernel(const float* __restrict__ x, half_t* __restrict__ out) {
    long idx = (long)blockIdx.x * 256 + threadIdx.x;
    if (idx >= (long)MTOK * CC) return;
    int k = (int)(idx % CC);
    int m = (int)(idx / CC);
    int b = m >> 8, n = m & 255;
    int ph = n >> 4, pw = n & 15;
    int c = k >> 8, pq = k & 255;
    int p = pq >> 4, q = pq & 15;
    float v = x[(((size_t)b*3 + c)*256 + (ph*16 + p))*256 + (pw*16 + q)];
    out[idx] = (half_t)v;
}

// ---------------------------------------------------------------- WMMA GEMM
// out[m,n] = sum_k A[m,k] * W[n,k] (+bias) (+GELU) (+residual). A,W f16, acc f32.
// 128x128 block tile, 256 threads = 8 waves (4 M-groups x 2 N-groups), 32x64/wave.
// Double-buffered LDS tiles filled by CDNA5 async DMA (global_load_async_to_lds_b128).
template<bool HAS_BIAS, bool DO_GELU, bool OUT16, bool RESID>
__global__ __launch_bounds__(256)
void gemm_wmma_kernel(const half_t* __restrict__ A, const half_t* __restrict__ W,
                      const float* __restrict__ bias,
                      float* __restrict__ outF, half_t* __restrict__ out16,
                      int M, int N, int K, int row_div, int row_stride)
{
    __shared__ __attribute__((aligned(32))) half_t As[2][128][40];
    __shared__ __attribute__((aligned(32))) half_t Wsh[2][128][40];

    const int tid  = threadIdx.x;
    const int lane = tid & 31;
    const int wave = tid >> 5;
    const int wm   = wave & 3;   // row wave group
    const int wn   = wave >> 2;  // col wave group
    const int mBlock = blockIdx.y * 128;
    const int nBlock = blockIdx.x * 128;
    const int lr = lane & 15;
    const int hs = (lane < 16) ? 0 : 1;

    v8f acc[2][4];
    for (int i = 0; i < 2; ++i)
        for (int j = 0; j < 4; ++j)
            acc[i][j] = (v8f){0.f,0.f,0.f,0.f,0.f,0.f,0.f,0.f};

    // per-thread async-copy source/dest (clamp OOB rows; stores are guarded later)
    const int ldrow = tid >> 1;
    const int ldoff = (tid & 1) << 4;
    int gm = mBlock + ldrow; if (gm >= M) gm = M - 1;
    int gn = nBlock + ldrow; if (gn >= N) gn = N - 1;
    const half_t* gA = A + (size_t)gm * K + ldoff;
    const half_t* gW = W + (size_t)gn * K + ldoff;

    const int nk = K >> 5;
    // prologue: issue tile 0
    {
        half_t* la = &As[0][ldrow][ldoff];
        half_t* lw = &Wsh[0][ldrow][ldoff];
        async_load_b128(gA,     la);
        async_load_b128(gA + 8, la + 8);
        async_load_b128(gW,     lw);
        async_load_b128(gW + 8, lw + 8);
    }

    for (int kt = 0; kt < nk; ++kt) {
        const int buf = kt & 1;
        if (kt + 1 < nk) {
            const int k1 = (kt + 1) << 5;
            half_t* la = &As[buf ^ 1][ldrow][ldoff];
            half_t* lw = &Wsh[buf ^ 1][ldrow][ldoff];
            async_load_b128(gA + k1,     la);
            async_load_b128(gA + k1 + 8, la + 8);
            async_load_b128(gW + k1,     lw);
            async_load_b128(gW + k1 + 8, lw + 8);
            async_wait4();          // current tile landed; next tile still in flight
        } else {
            async_wait0();
        }
        __syncthreads();

        v16h afrag[2], bfrag[4];
        for (int i = 0; i < 2; ++i) {
            int m = wm*32 + i*16 + lr;
            h8 lo = *(const h8*)(&As[buf][m][hs*8]);
            h8 hi = *(const h8*)(&As[buf][m][hs*8 + 16]);
            afrag[i] = combine16(lo, hi);
        }
        for (int j = 0; j < 4; ++j) {
            int n = wn*64 + j*16 + lr;
            h8 lo = *(const h8*)(&Wsh[buf][n][hs*8]);
            h8 hi = *(const h8*)(&Wsh[buf][n][hs*8 + 16]);
            bfrag[j] = combine16(lo, hi);
        }
        for (int i = 0; i < 2; ++i)
            for (int j = 0; j < 4; ++j)
                acc[i][j] = __builtin_amdgcn_wmma_f32_16x16x32_f16(
                    false, afrag[i], false, bfrag[j], (short)0, acc[i][j], false, false);
        __syncthreads();
    }

    // epilogue
    for (int i = 0; i < 2; ++i) {
        for (int j = 0; j < 4; ++j) {
            int n = nBlock + wn*64 + j*16 + lr;
            if (n >= N) continue;
            float bv = HAS_BIAS ? bias[n] : 0.0f;
            for (int r = 0; r < 8; ++r) {
                int m = mBlock + wm*32 + i*16 + ((lane < 16) ? r : 8 + r);
                if (m >= M) continue;
                float v = acc[i][j][r] + bv;
                if (DO_GELU) v = 0.5f * v * (1.0f + erff(v * 0.70710678118654752f));
                if (OUT16) {
                    out16[(size_t)m * N + n] = (half_t)v;
                } else {
                    long orow = (long)(m / row_div) * row_stride + (m % row_div);
                    size_t o = (size_t)orow * N + n;
                    if (RESID) v += outF[o];
                    outF[o] = v;
                }
            }
        }
    }
}

// ---------------------------------------------------------------- LayerNorm (wave per row)
template<typename OutT>
__global__ __launch_bounds__(256)
void ln_kernel(const float* __restrict__ in, const float* __restrict__ g,
               const float* __restrict__ beta, OutT* __restrict__ out,
               int rows, int row_div, int row_stride)
{
    int wave = threadIdx.x >> 5, lane = threadIdx.x & 31;
    int r = blockIdx.x * 8 + wave;
    if (r >= rows) return;
    long irow = (long)(r / row_div) * row_stride + (r % row_div);
    const float* x = in + irow * CC;
    float v[24];
    float s = 0.f;
    for (int e = 0; e < 24; ++e) { v[e] = x[lane + e*32]; s += v[e]; }
    for (int msk = 16; msk; msk >>= 1) s += __shfl_xor(s, msk, 32);
    float mu = s * (1.0f/CC);
    float vs = 0.f;
    for (int e = 0; e < 24; ++e) { float d = v[e]-mu; vs += d*d; }
    for (int msk = 16; msk; msk >>= 1) vs += __shfl_xor(vs, msk, 32);
    float rstd = rsqrtf(vs * (1.0f/CC) + 1e-5f);
    OutT* orow = out + (size_t)r * CC;
    for (int e = 0; e < 24; ++e) {
        int c = lane + e*32;
        orow[c] = (OutT)((v[e]-mu)*rstd*g[c] + beta[c]);
    }
}

// ---------------------------------------------------------------- patch LN + pos embed + ar token -> h
__global__ __launch_bounds__(256)
void embed_kernel(const float* __restrict__ patches, const float* __restrict__ pe_g,
                  const float* __restrict__ pe_b, const float* __restrict__ ar_token,
                  const float* __restrict__ aspect, const float* __restrict__ pos,
                  float* __restrict__ h)
{
    int wave = threadIdx.x >> 5, lane = threadIdx.x & 31;
    int r = blockIdx.x * 8 + wave;
    if (r >= MALL) return;
    int b = r / NALL, n = r % NALL;
    float* out = h + (size_t)r * CC;
    if (n == NTOK) {
        float sc = 1.0f + aspect[b] * 0.1f;
        for (int e = 0; e < 24; ++e) {
            int c = lane + e*32;
            out[c] = ar_token[c]*sc + pos[(size_t)NTOK*CC + c];
        }
    } else {
        const float* x = patches + ((size_t)b*NTOK + n) * CC;
        float v[24];
        float s = 0.f;
        for (int e = 0; e < 24; ++e) { v[e] = x[lane + e*32]; s += v[e]; }
        for (int msk = 16; msk; msk >>= 1) s += __shfl_xor(s, msk, 32);
        float mu = s * (1.0f/CC);
        float vs = 0.f;
        for (int e = 0; e < 24; ++e) { float d = v[e]-mu; vs += d*d; }
        for (int msk = 16; msk; msk >>= 1) vs += __shfl_xor(vs, msk, 32);
        float rstd = rsqrtf(vs * (1.0f/CC) + 1e-5f);
        for (int e = 0; e < 24; ++e) {
            int c = lane + e*32;
            out[c] = (v[e]-mu)*rstd*pe_g[c] + pe_b[c] + pos[(size_t)n*CC + c];
        }
    }
}

// ---------------------------------------------------------------- split qkv -> padded f16 K, transposed V
__global__ void prep_kv_kernel(const float* __restrict__ qkv,
                               half_t* __restrict__ k16, half_t* __restrict__ vT16)
{
    long idx = (long)blockIdx.x * 256 + threadIdx.x;
    if (idx >= (long)BB * NHD * NP * HD) return;
    int d  = (int)(idx & 63);
    int n  = (int)((idx >> 6) % NP);
    int bh = (int)(idx / ((long)NP * HD));
    int b  = bh / NHD, hh = bh % NHD;
    float kv = 0.f, vv = 0.f;
    if (n < NALL) {
        const float* base = qkv + ((size_t)(b*NALL + n))*(3*CC) + hh*HD + d;
        kv = base[CC];
        vv = base[2*CC];
    }
    k16[((size_t)bh*NP + n)*HD + d] = (half_t)kv;
    vT16[((size_t)bh*HD + d)*NP + n] = (half_t)vv;
}

// ---------------------------------------------------------------- global attention (WMMA)
// grid: (B*NH, 9 q-tiles of 32 rows), block 128 (4 waves)
__global__ __launch_bounds__(128)
void gattn_kernel(const float* __restrict__ qkv, const half_t* __restrict__ k16,
                  const half_t* __restrict__ vT16, half_t* __restrict__ out16)
{
    __shared__ __attribute__((aligned(32))) half_t q_s[32][72];
    __shared__ float S_s[32][296];
    __shared__ __attribute__((aligned(32))) half_t P_s[32][304];

    int bh = blockIdx.x;
    int b = bh / NHD, hh = bh % NHD;
    int qbase = blockIdx.y * 32;
    int tid = threadIdx.x, lane = tid & 31, wave = tid >> 5;
    const int lr = lane & 15;
    const int hs = (lane < 16) ? 0 : 1;

    // stage q (f32 -> f16)
    for (int idx = tid; idx < 32*HD; idx += 128) {
        int row = idx >> 6, d = idx & 63;
        int qrow = qbase + row;
        float qv = (qrow < NALL)
            ? qkv[((size_t)(b*NALL + qrow))*(3*CC) + hh*HD + d] : 0.0f;
        q_s[row][d] = (half_t)qv;
    }
    __syncthreads();

    const half_t* kb = k16 + (size_t)bh * NP * HD;
    // phase 1: S = Q K^T * scale (2 row-tiles x 18 col-tiles); two independent
    // accumulators so the two WMMAs per job have no RAW chain (no hazard NOPs).
    for (int job = wave; job < 36; job += 4) {
        int rt = job / 18, ct = job % 18;
        int m = rt*16 + lr;
        int key = ct*16 + lr;
        v8f acc0 = (v8f){0.f,0.f,0.f,0.f,0.f,0.f,0.f,0.f};
        v8f acc1 = acc0;
        {
            h8 lo = *(const h8*)(&q_s[m][hs*16]);
            h8 hi = *(const h8*)(&q_s[m][hs*16 + 8]);
            v16h a0 = combine16(lo, hi);
            v16h b0 = *(const v16h*)(kb + (size_t)key*HD + hs*16);
            lo = *(const h8*)(&q_s[m][32 + hs*16]);
            hi = *(const h8*)(&q_s[m][32 + hs*16 + 8]);
            v16h a1 = combine16(lo, hi);
            v16h b1 = *(const v16h*)(kb + (size_t)key*HD + 32 + hs*16);
            acc0 = __builtin_amdgcn_wmma_f32_16x16x32_f16(false, a0, false, b0, (short)0, acc0, false, false);
            acc1 = __builtin_amdgcn_wmma_f32_16x16x32_f16(false, a1, false, b1, (short)0, acc1, false, false);
        }
        for (int r = 0; r < 8; ++r) {
            int row = rt*16 + ((lane < 16) ? r : 8 + r);
            S_s[row][ct*16 + lr] = (acc0[r] + acc1[r]) * 0.125f;
        }
    }
    __syncthreads();

    // phase 2: softmax over 257 keys, write f16 probs (padded cols -> 0)
    if (tid < 32) {
        int row = tid;
        int qrow = qbase + row;
        if (qrow < NALL) {
            float mx = -1e30f;
            for (int j = 0; j < NALL; ++j) mx = fmaxf(mx, S_s[row][j]);
            float sum = 0.f;
            for (int j = 0; j < NALL; ++j) { float e = expf(S_s[row][j]-mx); S_s[row][j] = e; sum += e; }
            float inv = 1.0f / sum;
            for (int j = 0; j < NP; ++j)
                P_s[row][j] = (half_t)((j < NALL) ? S_s[row][j]*inv : 0.0f);
        } else {
            for (int j = 0; j < NP; ++j) P_s[row][j] = (half_t)0;
        }
    }
    __syncthreads();

    // phase 3: O = P V (2 row-tiles x 4 d-tiles); alternate two accumulators to
    // break the 9-deep WMMA dependency chain.
    const half_t* vb = vT16 + (size_t)bh * HD * NP;
    for (int job = wave; job < 8; job += 4) {
        int rt = job >> 2, dt = job & 3;
        int m = rt*16 + lr;
        int d = dt*16 + lr;
        v8f accA = (v8f){0.f,0.f,0.f,0.f,0.f,0.f,0.f,0.f};
        v8f accB = accA;
        for (int idx = 0; idx < 9; ++idx) {
            int kc = idx << 5;
            h8 lo = *(const h8*)(&P_s[m][kc + hs*16]);
            h8 hi = *(const h8*)(&P_s[m][kc + hs*16 + 8]);
            v16h a = combine16(lo, hi);
            v16h bf = *(const v16h*)(vb + (size_t)d*NP + kc + hs*16);
            if (idx & 1)
                accB = __builtin_amdgcn_wmma_f32_16x16x32_f16(false, a, false, bf, (short)0, accB, false, false);
            else
                accA = __builtin_amdgcn_wmma_f32_16x16x32_f16(false, a, false, bf, (short)0, accA, false, false);
        }
        for (int r = 0; r < 8; ++r) {
            int qrow = qbase + rt*16 + ((lane < 16) ? r : 8 + r);
            if (qrow < NALL)
                out16[((size_t)(b*NALL + qrow))*CC + hh*HD + dt*16 + lr] = (half_t)(accA[r] + accB[r]);
        }
    }
}

// ---------------------------------------------------------------- local 7x7-window attention (scalar, ~1% of FLOPs)
__global__ __launch_bounds__(64)
void lattn_kernel(const float* __restrict__ qkv, half_t* __restrict__ out16)
{
    __shared__ float aw[49];
    int gid = blockIdx.x;             // b*12*256 + h*256 + n
    int n  = gid & 255;
    int hh = (gid >> 8) % NHD;
    int b  = gid / (NHD * 256);
    int qh = n >> 4, qw = n & 15;
    int t = threadIdx.x;
    const size_t RS = 3*CC;
    const float* qp = qkv + ((size_t)(b*NTOK + n))*RS + hh*HD;

    if (t < 49) {
        int dh = t / 7, dw = t % 7;
        int kh = qh + dh - 3, kw = qw + dw - 3;
        float sc = 0.0f;   // reference zero-pads K -> OOB score is 0, kept in softmax
        if (kh >= 0 && kh < 16 && kw >= 0 && kw < 16) {
            const float* kp = qkv + ((size_t)(b*NTOK + kh*16 + kw))*RS + CC + hh*HD;
            float dot = 0.f;
            for (int d = 0; d < HD; ++d) dot += qp[d]*kp[d];
            sc = dot * 0.125f;
        }
        aw[t] = sc;
    }
    __syncthreads();
    if (t == 0) {
        float mx = aw[0];
        for (int i = 1; i < 49; ++i) mx = fmaxf(mx, aw[i]);
        float sum = 0.f;
        for (int i = 0; i < 49; ++i) { float e = expf(aw[i]-mx); aw[i] = e; sum += e; }
        float inv = 1.0f/sum;
        for (int i = 0; i < 49; ++i) aw[i] *= inv;
    }
    __syncthreads();
    float o = 0.f;
    for (int i = 0; i < 49; ++i) {
        int dh = i / 7, dw = i % 7;
        int kh = qh + dh - 3, kw = qw + dw - 3;
        if (kh >= 0 && kh < 16 && kw >= 0 && kw < 16)
            o += aw[i] * qkv[((size_t)(b*NTOK + kh*16 + kw))*RS + 2*CC + hh*HD + t];
    }
    out16[((size_t)(b*NTOK + n))*CC + hh*HD + t] = (half_t)o;
}

// ---------------------------------------------------------------- host
extern "C" void kernel_launch(void* const* d_in, const int* in_sizes, int n_in,
                              void* d_out, int out_size, void* d_ws, size_t ws_size,
                              hipStream_t stream)
{
    const float* x        = (const float*)d_in[0];
    const float* aspect   = (const float*)d_in[1];
    const float* conv_w   = (const float*)d_in[2];
    const float* conv_b   = (const float*)d_in[3];
    const float* pe_g     = (const float*)d_in[4];
    const float* pe_b     = (const float*)d_in[5];
    const float* ar_token = (const float*)d_in[6];
    const float* pos      = (const float*)d_in[7];
    const float* ln1_g    = (const float*)d_in[8];
    const float* ln1_b    = (const float*)d_in[9];
    const float* ln2_g    = (const float*)d_in[10];
    const float* ln2_b    = (const float*)d_in[11];
    const float* loc_qkv_w  = (const float*)d_in[12];
    const float* loc_proj_w = (const float*)d_in[13];
    const float* loc_proj_b = (const float*)d_in[14];
    const float* glob_in_w  = (const float*)d_in[15];
    const float* glob_in_b  = (const float*)d_in[16];
    const float* glob_out_w = (const float*)d_in[17];
    const float* glob_out_b = (const float*)d_in[18];
    const float* mlp_w1   = (const float*)d_in[19];
    const float* mlp_b1   = (const float*)d_in[20];
    const float* mlp_w2   = (const float*)d_in[21];
    const float* mlp_b2   = (const float*)d_in[22];
    const float* fin_g    = (const float*)d_in[23];
    const float* fin_b    = (const float*)d_in[24];
    float* out = (float*)d_out;

    char* ws = (char*)d_ws;
    size_t off = 0;
    auto alloc = [&](size_t bytes) -> char* {
        char* p = ws + off;
        off = (off + bytes + 255) & ~(size_t)255;
        return p;
    };

    const long N_CONV = 589824;           // 768*768
    const long N_LQKV = 6L*3*CC*CC;
    const long N_LPRJ = 6L*CC*CC;
    const long N_GIN  = 6L*3*CC*CC;
    const long N_GOUT = 6L*CC*CC;
    const long N_M1   = 12L*MLPD*CC;
    const long N_M2   = 12L*CC*MLPD;

    half_t* wc16  = (half_t*)alloc((size_t)N_CONV*2);
    half_t* wlq16 = (half_t*)alloc((size_t)N_LQKV*2);
    half_t* wlp16 = (half_t*)alloc((size_t)N_LPRJ*2);
    half_t* wgi16 = (half_t*)alloc((size_t)N_GIN*2);
    half_t* wgo16 = (half_t*)alloc((size_t)N_GOUT*2);
    half_t* wm116 = (half_t*)alloc((size_t)N_M1*2);
    half_t* wm216 = (half_t*)alloc((size_t)N_M2*2);
    half_t* a16   = (half_t*)alloc((size_t)MTOK*CC*2);     // im2col
    float*  patch = (float* )alloc((size_t)MTOK*CC*4);
    float*  h     = (float* )alloc((size_t)MALL*CC*4);
    half_t* x16   = (half_t*)alloc((size_t)MALL*CC*2);
    float*  qkv   = (float* )alloc((size_t)MALL*3*CC*4);
    half_t* at16  = (half_t*)alloc((size_t)MALL*CC*2);
    half_t* m116  = (half_t*)alloc((size_t)MALL*MLPD*2);
    half_t* k16   = (half_t*)alloc((size_t)BB*NHD*NP*HD*2);
    half_t* vT16  = (half_t*)alloc((size_t)BB*NHD*HD*NP*2);
    (void)ws_size; (void)in_sizes; (void)n_in; (void)out_size;

    auto cvt = [&](const float* src, half_t* dst, long n) {
        cvt_kernel<<<(int)((n + 255)/256), 256, 0, stream>>>(src, dst, n);
    };
    cvt(conv_w,     wc16,  N_CONV);
    cvt(loc_qkv_w,  wlq16, N_LQKV);
    cvt(loc_proj_w, wlp16, N_LPRJ);
    cvt(glob_in_w,  wgi16, N_GIN);
    cvt(glob_out_w, wgo16, N_GOUT);
    cvt(mlp_w1,     wm116, N_M1);
    cvt(mlp_w2,     wm216, N_M2);

    const int IDIV = 1 << 30;   // identity remap
    const int GB_TOK = (MTOK + 127)/128, GB_ALL = (MALL + 127)/128;
    const int NB_C = CC/128, NB_3C = 3*CC/128, NB_M = MLPD/128;

    // patch embedding: im2col + GEMM(conv_w, conv_b)
    im2col_kernel<<<(int)(((long)MTOK*CC + 255)/256), 256, 0, stream>>>(x, a16);
    gemm_wmma_kernel<true,false,false,false><<<dim3(NB_C, GB_TOK), 256, 0, stream>>>(
        a16, wc16, conv_b, patch, nullptr, MTOK, CC, CC, IDIV, 1);
    embed_kernel<<<(MALL + 7)/8, 256, 0, stream>>>(patch, pe_g, pe_b, ar_token, aspect, pos, h);

    for (int i = 0; i < 12; ++i) {
        if ((i & 1) == 0) {
            int li = i >> 1;
            ln_kernel<half_t><<<(MTOK + 7)/8, 256, 0, stream>>>(
                h, ln1_g + (size_t)i*CC, ln1_b + (size_t)i*CC, x16, MTOK, NTOK, NALL);
            gemm_wmma_kernel<false,false,false,false><<<dim3(NB_3C, GB_TOK), 256, 0, stream>>>(
                x16, wlq16 + (size_t)li*3*CC*CC, nullptr, qkv, nullptr, MTOK, 3*CC, CC, IDIV, 1);
            lattn_kernel<<<BB*NHD*256, 64, 0, stream>>>(qkv, at16);
            gemm_wmma_kernel<true,false,false,true><<<dim3(NB_C, GB_TOK), 256, 0, stream>>>(
                at16, wlp16 + (size_t)li*CC*CC, loc_proj_b + (size_t)li*CC,
                h, nullptr, MTOK, CC, CC, NTOK, NALL);
        } else {
            int gi = i >> 1;
            ln_kernel<half_t><<<(MALL + 7)/8, 256, 0, stream>>>(
                h, ln1_g + (size_t)i*CC, ln1_b + (size_t)i*CC, x16, MALL, IDIV, 1);
            gemm_wmma_kernel<true,false,false,false><<<dim3(NB_3C, GB_ALL), 256, 0, stream>>>(
                x16, wgi16 + (size_t)gi*3*CC*CC, glob_in_b + (size_t)gi*3*CC,
                qkv, nullptr, MALL, 3*CC, CC, IDIV, 1);
            prep_kv_kernel<<<(int)(((long)BB*NHD*NP*HD + 255)/256), 256, 0, stream>>>(qkv, k16, vT16);
            gattn_kernel<<<dim3(BB*NHD, 9), 128, 0, stream>>>(qkv, k16, vT16, at16);
            gemm_wmma_kernel<true,false,false,true><<<dim3(NB_C, GB_ALL), 256, 0, stream>>>(
                at16, wgo16 + (size_t)gi*CC*CC, glob_out_b + (size_t)gi*CC,
                h, nullptr, MALL, CC, CC, IDIV, 1);
        }
        ln_kernel<half_t><<<(MALL + 7)/8, 256, 0, stream>>>(
            h, ln2_g + (size_t)i*CC, ln2_b + (size_t)i*CC, x16, MALL, IDIV, 1);
        gemm_wmma_kernel<true,true,true,false><<<dim3(NB_M, GB_ALL), 256, 0, stream>>>(
            x16, wm116 + (size_t)i*MLPD*CC, mlp_b1 + (size_t)i*MLPD,
            nullptr, m116, MALL, MLPD, CC, IDIV, 1);
        gemm_wmma_kernel<true,false,false,true><<<dim3(NB_C, GB_ALL), 256, 0, stream>>>(
            m116, wm216 + (size_t)i*CC*MLPD, mlp_b2 + (size_t)i*CC,
            h, nullptr, MALL, CC, MLPD, IDIV, 1);
    }

    ln_kernel<float><<<(MALL + 7)/8, 256, 0, stream>>>(h, fin_g, fin_b, out, MALL, IDIV, 1);
}
